// AttnBlock_20306605375782
// MI455X (gfx1250) — matrix operand order repaired and checked by reading, not compiled, not run on hardware
//
#include <hip/hip_runtime.h>
#include <hip/hip_bf16.h>
#include <math.h>

// ---------------------------------------------------------------------------
// AttnBlock (B=4, C=512, H=W=64, N=4096) for gfx1250 (MI455X), wave32 + WMMA.
//
// Pipeline (all GEMMs are the same "NT" WMMA kernel: D[m][n] = sum_k A[m][k]*B[n][k]):
//   1) groupnorm          : x(B,C,N) f32 -> hT(B,N,C) f16
//   2) f32->f16 weights   : wq,wk,wv,wp (C,C) row-major [o][c]
//   3) qT = hT x Wq^T + bq   (N x C)   f16
//      kT = hT x Wk^T + bk             f16
//      v  = Wv x hT^T + bv  (C x N)    f16
//   4) S  = qT x kT^T * C^-0.5 (N x N) f32
//   5) softmax rows of S, write P as f16 in-place (lda=8192)
//   6) OT = P x v^T          (N x C)   f16
//   7) out = Wp x OT^T + bp + residual (C x N) f32
//
// GEMM K-loop is double-buffered through LDS using CDNA5 async copies
// (global_load_async_to_lds_b128 + s_wait_asynccnt), so HBM/L2 fetch of
// tile k+1 overlaps the WMMAs on tile k.
// ---------------------------------------------------------------------------

typedef _Float16 half16 __attribute__((ext_vector_type(16)));
typedef _Float16 half8  __attribute__((ext_vector_type(8)));
typedef float    floatx8 __attribute__((ext_vector_type(8)));

#define BM 128
#define BN 128
#define BK 64
#define LDT (BK + 8)            // padded LDS row stride in halves (144B)
#define TILE_HALFS (BM * LDT)   // halves per buffer
#define TILE_BYTES (TILE_HALFS * 2)

// ---------------------------------------------------------------------------
// Generic NT WMMA GEMM: D[m][n] = scale * (sum_k A[m][k] * B[n][k]) [+bias][+res]
//   A: M x K rows (lda), B: N x K rows (ldb), D: M x N row-major. grid.z = batch.
//   BIAS_MODE: 0 = none, 1 = bias[m], 2 = bias[n].
// 256 threads = 8 waves arranged 4(M) x 2(N); wave tile 32x64 = 2x4 WMMA tiles.
// ---------------------------------------------------------------------------
template <typename OutT, int BIAS_MODE, bool RESIDUAL>
__global__ __launch_bounds__(256) void gemm_nt_wmma(
    const _Float16* __restrict__ A, long long strideA, int lda,
    const _Float16* __restrict__ B, long long strideB, int ldb,
    OutT* __restrict__ D, long long strideD,
    const float* __restrict__ bias,
    const float* __restrict__ residual, long long strideR,
    int M, int N, int K, float scale)
{
    __shared__ __attribute__((aligned(32))) _Float16 As[2][TILE_HALFS];
    __shared__ __attribute__((aligned(32))) _Float16 Bs[2][TILE_HALFS];

    const int tid  = threadIdx.x;
    const int wave = tid >> 5;
    const int lane = tid & 31;
    const int l    = lane & 15;
    const int h    = lane >> 4;          // half-wave select per ISA WMMA layouts

    const int wm = wave >> 1;            // 0..3 -> 32-row strip
    const int wn = wave & 1;             // 0..1 -> 64-col strip
    const int m0 = blockIdx.y * BM;
    const int n0 = blockIdx.x * BN;
    const int z  = blockIdx.z;

    A += (long long)z * strideA;
    B += (long long)z * strideB;
    D += (long long)z * strideD;

    // --- per-thread async-copy addressing: 4 chunks of 16B per operand tile ---
    // chunk it: row = tid/8 + it*32, col = (tid%8)*8 halves
    const int colh = (tid & 7) * 8;
    unsigned ldsA[4], ldsB[4];
    unsigned long long gA[4], gB[4];
    #pragma unroll
    for (int it = 0; it < 4; ++it) {
        const int row = (tid >> 3) + it * 32;
        ldsA[it] = (unsigned)(unsigned long long)&As[0][row * LDT + colh];
        ldsB[it] = (unsigned)(unsigned long long)&Bs[0][row * LDT + colh];
        gA[it] = (unsigned long long)(const void*)&A[(long long)(m0 + row) * lda + colh];
        gB[it] = (unsigned long long)(const void*)&B[(long long)(n0 + row) * ldb + colh];
    }

    auto issue_tile = [&](int bufsel, int k0) {
        const unsigned bo = bufsel ? (unsigned)TILE_BYTES : 0u;
        const unsigned long long kb = (unsigned long long)k0 * 2;  // byte offset along K
        #pragma unroll
        for (int it = 0; it < 4; ++it) {
            asm volatile("global_load_async_to_lds_b128 %0, %1, off"
                         :: "v"(ldsA[it] + bo), "v"(gA[it] + kb) : "memory");
            asm volatile("global_load_async_to_lds_b128 %0, %1, off"
                         :: "v"(ldsB[it] + bo), "v"(gB[it] + kb) : "memory");
        }
    };

    floatx8 acc[2][4];
    #pragma unroll
    for (int i = 0; i < 2; ++i)
        #pragma unroll
        for (int j = 0; j < 4; ++j)
            #pragma unroll
            for (int e = 0; e < 8; ++e) acc[i][j][e] = 0.0f;

    // prologue: start tile 0 into buffer 0
    issue_tile(0, 0);

    int bufi = 0;
    for (int k0 = 0; k0 < K; k0 += BK) {
        const bool has_next = (k0 + BK) < K;
        if (has_next) {
            issue_tile(bufi ^ 1, k0 + BK);
            // async loads complete in order: cnt<=8 => current tile's 8 are done,
            // next tile's 8 still in flight behind the WMMAs.
            asm volatile("s_wait_asynccnt 0x8" ::: "memory");
        } else {
            asm volatile("s_wait_asynccnt 0x0" ::: "memory");
        }
        __syncthreads();   // current tile visible to all waves

        const _Float16* as = As[bufi];
        const _Float16* bs = Bs[bufi];
        #pragma unroll
        for (int ks = 0; ks < BK; ks += 32) {
            half16 afrag[2], bfrag[4];
            // A fragment (16x32 f16): lane(l,h): K = {h*8..+7, 16+h*8..+7}
            #pragma unroll
            for (int i = 0; i < 2; ++i) {
                const _Float16* ap = &as[(wm * 32 + i * 16 + l) * LDT + ks];
                ((half8*)&afrag[i])[0] = *(const half8*)(ap + h * 8);
                ((half8*)&afrag[i])[1] = *(const half8*)(ap + 16 + h * 8);
            }
            // B fragment (32x16 f16): lane(l,h): column n=l, K = {h*8..+7, 16+h*8..+7}
            #pragma unroll
            for (int j = 0; j < 4; ++j) {
                const _Float16* bp = &bs[(wn * 64 + j * 16 + l) * LDT + ks];
                ((half8*)&bfrag[j])[0] = *(const half8*)(bp + h * 8);
                ((half8*)&bfrag[j])[1] = *(const half8*)(bp + 16 + h * 8);
            }
            #pragma unroll
            for (int i = 0; i < 2; ++i)
                #pragma unroll
                for (int j = 0; j < 4; ++j)
                    acc[i][j] = __builtin_amdgcn_wmma_f32_16x16x32_f16(
                        false, afrag[i], false, bfrag[j],
                        (short)0, acc[i][j], false, false);
        }
        __syncthreads();   // all waves done reading bufi before it is refilled
        bufi ^= 1;
    }

    // Epilogue. C/D layout: lane(l,h), vgpr r -> (m = h*8 + r, n = l).
    #pragma unroll
    for (int i = 0; i < 2; ++i) {
        #pragma unroll
        for (int j = 0; j < 4; ++j) {
            const int mg0 = m0 + wm * 32 + i * 16 + h * 8;
            const int ng  = n0 + wn * 64 + j * 16 + l;
            #pragma unroll
            for (int r = 0; r < 8; ++r) {
                const int mg = mg0 + r;
                float val = acc[i][j][r] * scale;
                if (BIAS_MODE == 1) val += bias[mg];
                if (BIAS_MODE == 2) val += bias[ng];
                if (RESIDUAL)
                    val += residual[(long long)z * strideR + (long long)mg * N + ng];
                D[(long long)mg * N + ng] = (OutT)val;
            }
        }
    }
}

// ---------------------------------------------------------------------------
// GroupNorm: x(B,C,N) f32 -> hT(B,N,C) f16.  One block per (batch, group).
// Group = 16 channels x 4096 spatial.
// ---------------------------------------------------------------------------
__global__ __launch_bounds__(256) void groupnorm_kernel(
    const float* __restrict__ x, const float* __restrict__ gamma,
    const float* __restrict__ beta, _Float16* __restrict__ hT)
{
    const int bg = blockIdx.x;
    const int b = bg >> 5, g = bg & 31;
    const int tid = threadIdx.x;
    const float* xg = x + ((long long)b * 512 + g * 16) * 4096;
    _Float16* ht = hT + (long long)b * 4096 * 512 + g * 16;

    float s = 0.f, ss = 0.f;
    for (int idx = tid; idx < 16 * 4096; idx += 256) {
        float v = xg[idx];
        s += v;
        ss += v * v;
    }
    __shared__ float r1[256], r2[256];
    r1[tid] = s; r2[tid] = ss;
    __syncthreads();
    for (int off = 128; off > 0; off >>= 1) {
        if (tid < off) { r1[tid] += r1[tid + off]; r2[tid] += r2[tid + off]; }
        __syncthreads();
    }
    const float mean = r1[0] * (1.0f / 65536.0f);
    const float var  = r2[0] * (1.0f / 65536.0f) - mean * mean;
    const float inv  = rsqrtf(var + 1e-6f);

    float gm[16], bt[16];
    #pragma unroll
    for (int c = 0; c < 16; ++c) {
        gm[c] = gamma[g * 16 + c] * inv;
        bt[c] = beta[g * 16 + c];
    }
    // one pixel per thread-iteration: 16 coalesced strided reads, one 32B write
    for (int n = tid; n < 4096; n += 256) {
        half16 o;
        #pragma unroll
        for (int c = 0; c < 16; ++c)
            o[c] = (_Float16)((xg[c * 4096 + n] - mean) * gm[c] + bt[c]);
        *(half16*)&ht[(long long)n * 512] = o;
    }
}

// ---------------------------------------------------------------------------
// fp32 -> fp16 weight conversion
// ---------------------------------------------------------------------------
__global__ __launch_bounds__(256) void f32_to_f16_kernel(
    const float* __restrict__ s, _Float16* __restrict__ d, int n)
{
    const int i = blockIdx.x * 256 + threadIdx.x;
    if (i < n) d[i] = (_Float16)s[i];
}

// ---------------------------------------------------------------------------
// Row softmax over S (f32, row length 4096), writing P as f16 IN-PLACE into the
// start of each f32 row (so P has row stride 8192 halves).
// grid = (4096 rows, B batches), block = 256.
// ---------------------------------------------------------------------------
__global__ __launch_bounds__(256) void softmax_kernel(float* __restrict__ S)
{
    const long long row = (long long)blockIdx.y * gridDim.x + blockIdx.x;
    float* sr = S + row * 4096;
    const int tid = threadIdx.x;

    float v[16];
    float m = -3.4e38f;
    #pragma unroll
    for (int p = 0; p < 16; ++p) {
        v[p] = sr[tid + p * 256];
        m = fmaxf(m, v[p]);
    }
    __shared__ float red[256];
    red[tid] = m;
    __syncthreads();
    for (int off = 128; off > 0; off >>= 1) {
        if (tid < off) red[tid] = fmaxf(red[tid], red[tid + off]);
        __syncthreads();
    }
    m = red[0];
    __syncthreads();

    float s = 0.f;
    #pragma unroll
    for (int p = 0; p < 16; ++p) {
        v[p] = __expf(v[p] - m);
        s += v[p];
    }
    red[tid] = s;
    __syncthreads();
    for (int off = 128; off > 0; off >>= 1) {
        if (tid < off) red[tid] += red[tid + off];
        __syncthreads();
    }
    const float inv = 1.0f / red[0];

    _Float16* pr = (_Float16*)sr;   // in-place: all reads already in registers
    #pragma unroll
    for (int p = 0; p < 16; ++p)
        pr[tid + p * 256] = (_Float16)(v[p] * inv);
}

// ---------------------------------------------------------------------------
// Host-side orchestration
// ---------------------------------------------------------------------------
extern "C" void kernel_launch(void* const* d_in, const int* in_sizes, int n_in,
                              void* d_out, int out_size, void* d_ws, size_t ws_size,
                              hipStream_t stream)
{
    (void)in_sizes; (void)n_in; (void)out_size; (void)ws_size;

    const float* x    = (const float*)d_in[0];   // (B,C,H,W) = (4,512,4096)
    const float* gns  = (const float*)d_in[1];
    const float* gnb  = (const float*)d_in[2];
    const float* wq   = (const float*)d_in[3];
    const float* bq   = (const float*)d_in[4];
    const float* wk   = (const float*)d_in[5];
    const float* bk   = (const float*)d_in[6];
    const float* wv   = (const float*)d_in[7];
    const float* bv   = (const float*)d_in[8];
    const float* wp   = (const float*)d_in[9];
    const float* bp   = (const float*)d_in[10];
    float* out = (float*)d_out;

    const long long NT = 4096, C = 512;
    const long long BNC = 4LL * NT * C;           // per-tensor f16 element count

    char* ws = (char*)d_ws;
    _Float16* hT = (_Float16*)ws;  ws += BNC * 2;                 // (B,N,C)
    _Float16* qT = (_Float16*)ws;  ws += BNC * 2;                 // (B,N,C)
    _Float16* kT = (_Float16*)ws;  ws += BNC * 2;                 // (B,N,C)
    _Float16* vv = (_Float16*)ws;  ws += BNC * 2;                 // (B,C,N)
    _Float16* OT = (_Float16*)ws;  ws += BNC * 2;                 // (B,N,C)
    _Float16* wq16 = (_Float16*)ws; ws += C * C * 2;
    _Float16* wk16 = (_Float16*)ws; ws += C * C * 2;
    _Float16* wv16 = (_Float16*)ws; ws += C * C * 2;
    _Float16* wp16 = (_Float16*)ws; ws += C * C * 2;
    float* S = (float*)ws;                                        // (B,N,N) f32; P f16 in-place

    // 1) GroupNorm -> hT
    groupnorm_kernel<<<dim3(4 * 32), dim3(256), 0, stream>>>(x, gns, gnb, hT);

    // 2) weight downcast
    const int WN = (int)(C * C);
    f32_to_f16_kernel<<<dim3((WN + 255) / 256), dim3(256), 0, stream>>>(wq, wq16, WN);
    f32_to_f16_kernel<<<dim3((WN + 255) / 256), dim3(256), 0, stream>>>(wk, wk16, WN);
    f32_to_f16_kernel<<<dim3((WN + 255) / 256), dim3(256), 0, stream>>>(wv, wv16, WN);
    f32_to_f16_kernel<<<dim3((WN + 255) / 256), dim3(256), 0, stream>>>(wp, wp16, WN);

    const long long sNC = NT * C;        // per-batch (N,C)/(C,N) stride
    const long long sNN = NT * NT;       // per-batch S f32 stride
    const long long sP  = NT * 8192;     // per-batch P stride in halves (aliased in S)

    // 3) qT = hT x Wq^T + bq ; kT = hT x Wk^T + bk   (M=4096, N=512, K=512)
    gemm_nt_wmma<_Float16, 2, false><<<dim3(512 / BN, 4096 / BM, 4), dim3(256), 0, stream>>>(
        hT, sNC, 512, wq16, 0, 512, qT, sNC, bq, nullptr, 0, 4096, 512, 512, 1.0f);
    gemm_nt_wmma<_Float16, 2, false><<<dim3(512 / BN, 4096 / BM, 4), dim3(256), 0, stream>>>(
        hT, sNC, 512, wk16, 0, 512, kT, sNC, bk, nullptr, 0, 4096, 512, 512, 1.0f);
    //    v = Wv x hT^T + bv   (M=512, N=4096, K=512)
    gemm_nt_wmma<_Float16, 1, false><<<dim3(4096 / BN, 512 / BM, 4), dim3(256), 0, stream>>>(
        wv16, 0, 512, hT, sNC, 512, vv, sNC, bv, nullptr, 0, 512, 4096, 512, 1.0f);

    // 4) S = qT x kT^T * C^-0.5   (M=N=4096, K=512), f32 out
    const float inv_sqrt_c = 0.044194173824159216f;   // 512^-0.5
    gemm_nt_wmma<float, 0, false><<<dim3(4096 / BN, 4096 / BM, 4), dim3(256), 0, stream>>>(
        qT, sNC, 512, kT, sNC, 512, S, sNN, nullptr, nullptr, 0, 4096, 4096, 512, inv_sqrt_c);

    // 5) softmax rows, P written f16 in-place (row stride 8192 halves)
    softmax_kernel<<<dim3(4096, 4), dim3(256), 0, stream>>>(S);

    // 6) OT = P x v^T   (M=4096, N=512, K=4096), A = P with lda = 8192
    gemm_nt_wmma<_Float16, 0, false><<<dim3(512 / BN, 4096 / BM, 4), dim3(256), 0, stream>>>(
        (const _Float16*)S, sP, 8192, vv, sNC, 4096, OT, sNC, nullptr, nullptr, 0,
        4096, 512, 4096, 1.0f);

    // 7) out = Wp x OT^T + bp + residual  (M=512, N=4096, K=512), f32 out
    gemm_nt_wmma<float, 1, true><<<dim3(4096 / BN, 512 / BM, 4), dim3(256), 0, stream>>>(
        wp16, 0, 512, OT, sNC, 512, out, sNC, bp, x, sNC, 512, 4096, 512, 1.0f);
}
